// MGCN_3058016714957
// MI455X (gfx1250) — matrix operand order repaired and checked by reading, not compiled, not run on hardware
//
#include <hip/hip_runtime.h>
#include <hip/hip_bf16.h>

// MGCN fused kernel for gfx1250 (MI455X).
// out[b,i,g] = A[i,i]*M[i,g]*(h0-h1)[b,i,g] + sum_j Asym[i,j]*(M[j,g]*h1[b,j,g]) + bias[g]
// where h0 = x@W0, h1 = x@W1, Asym = 0.5*((adj+adj2) + (adj+adj2)^T).
//
// Per workgroup: 16 batches (272 rows) x 128 cols. 8 waves; wave w owns cols 16w..16w+15.
// Stage 1: f16 WMMA GEMM (f32 accum) from LDS-staged x and register-held W tiles.
// Stage 2: 17x17 j-aggregation as one v_wmma_f32_16x16x32_f16 per tile, using a
//          zero-padded wide Asym row in LDS so block masking costs zero VALU.
// All b/j row indices are folded to compile-time constants (selected by lane-half),
// eliminating the div-by-17 chains of the previous revision.

typedef __attribute__((ext_vector_type(16))) _Float16 v16h;
typedef __attribute__((ext_vector_type(8)))  _Float16 h8;
typedef __attribute__((ext_vector_type(4)))  _Float16 h4;
typedef __attribute__((ext_vector_type(8)))  float    v8f;
typedef __attribute__((ext_vector_type(4)))  float    f4;

#define TB        16           // batches per workgroup
#define ROWS      (TB * 17)    // 272 rows per workgroup
#define XSTRIDE   136          // halves per row in xsh (128 + 8 pad, keeps 16B align + bank rotation)
#define ZSTRIDE   296          // halves per column in zsh (288 rows used + pad, 16B align + bank rotation)

__global__ __launch_bounds__(256)
void mgcn_fused_wmma(const float* __restrict__ x,
                     const float* __restrict__ W,
                     const float* __restrict__ Mmat,
                     const float* __restrict__ adj,
                     const float* __restrict__ adj2,
                     const float* __restrict__ bias,
                     float* __restrict__ out) {
    __shared__ __align__(16) _Float16 xsh[ROWS * XSTRIDE];   // x tile, f16 row-major
    __shared__ __align__(16) _Float16 zsh[128 * ZSTRIDE];    // z1 = M*h1, f16, [g][row]
    __shared__ __align__(16) float    Msh[17 * 128];         // M, f32
    __shared__ __align__(16) _Float16 Af16w[17 * 64];        // Asym row j at [j*64+24..+40], rest 0
    __shared__ __align__(16) float    adiag[17];             // Asym[j][j]
    __shared__ __align__(16) float    aL[17 * 2];            // Asym[j][15], Asym[j][16]

    const int tid = threadIdx.x;
    const int l   = tid & 31;          // lane in wave32
    const int wv  = tid >> 5;          // wave 0..7
    const int n   = l & 15;
    const bool hi = (l >= 16);
    const int g   = wv * 16 + n;       // this lane's output column

    const float* __restrict__ xg = x + (size_t)blockIdx.x * ROWS * 128;
    float* __restrict__ outp     = out + (size_t)blockIdx.x * ROWS * 128;
    const float* __restrict__ W0 = W;
    const float* __restrict__ W1 = W + 128 * 128;

    // ---- W0/W1 B-operands into registers (WMMA B layout:
    //      lane l holds col n=l%16, half e -> K = (e&7) + (e>=8 ? 16 : 0) + (l>=16 ? 8 : 0) + 32*kt). ----
    v16h w0t[4], w1t[4];
#pragma unroll
    for (int kt = 0; kt < 4; ++kt) {
        v16h a0 = {}, a1 = {};
#pragma unroll
        for (int e = 0; e < 16; ++e) {
            int f = kt * 32 + (e & 7) + (e >= 8 ? 16 : 0) + (hi ? 8 : 0);
            a0[e] = (_Float16)W0[f * 128 + g];
            a1[e] = (_Float16)W1[f * 128 + g];
        }
        w0t[kt] = a0; w1t[kt] = a1;
    }
    const float biasv = bias[g];

    // ---- stage x tile: global f32 -> LDS f16 (coalesced b128 loads, b64 DS stores) ----
    for (int c = tid; c < ROWS * 32; c += 256) {
        int row = c >> 5, kc = c & 31;
        f4 v = ((const f4*)xg)[c];
        h4 hv;
        hv[0] = (_Float16)v[0]; hv[1] = (_Float16)v[1];
        hv[2] = (_Float16)v[2]; hv[3] = (_Float16)v[3];
        *(h4*)&xsh[row * XSTRIDE + kc * 4] = hv;
    }
    // ---- stage M ----
    for (int i = tid; i < 17 * 128; i += 256) Msh[i] = Mmat[i];
    // ---- symmetrized A: zero-padded wide f16 rows + diag/leftover f32 tables ----
    for (int i = tid; i < 17 * 64; i += 256) {
        int r = i >> 6, slot = i & 63;
        _Float16 hv = (_Float16)0.0f;
        if (slot >= 24 && slot <= 40) {
            int c = slot - 24;
            float v = 0.5f * (adj[r * 17 + c] + adj2[r * 17 + c] +
                              adj[c * 17 + r] + adj2[c * 17 + r]);
            hv = (_Float16)v;
            if (c == r)  adiag[r]     = v;
            if (c == 15) aL[r * 2 + 0] = v;
            if (c == 16) aL[r * 2 + 1] = v;
        }
        Af16w[i] = hv;
    }
    // ---- zero z1 tail rows 272..287 (read by edge-tile K windows) ----
    {
        int gg = tid >> 1, p = tid & 1;
        *(h8*)&zsh[gg * ZSTRIDE + ROWS + p * 8] = (h8){};
    }
    __syncthreads();   // the only barrier: z1 traffic is wave-local afterwards

    // ---- stage 2: j-aggregation of out-tile u via one WMMA + 2-row patch ----
    auto stage2 = [&](int u, const v8f& pc) {
        const int R0  = 16 * u;            // constants after unroll
        const int b0  = R0 / 17;
        const int s   = 17 * b0;
        const int thr = s + 17 - R0;       // lane n >= thr => out-row is batch b0+1
        // B operand: z1 rows s..s+31 for this lane's column (u16 loads, imm offsets)
        v16h bz = {};
        const int zb = g * ZSTRIDE + s + (hi ? 8 : 0);
#pragma unroll
        for (int e = 0; e < 16; ++e) bz[e] = zsh[zb + e + (e >= 8 ? 8 : 0)];
        // A operand: pure loads from zero-padded wide row (mask encoded in padding)
        const int dB = (thr <= 15) ? ((n >= thr) ? 1 : 0) : 0;
        const int jm = n + (R0 - s) - 17 * dB;
        const int abase = jm * 64 + 24 + (hi ? 8 : 0) - 17 * dB;
        v16h abig = {};
#pragma unroll
        for (int e = 0; e < 16; ++e) {
            int Ke = e + (e >= 8 ? 8 : 0);
            abig[e] = Af16w[abase + Ke];
        }
        v8f d = __builtin_amdgcn_wmma_f32_16x16x32_f16(false, abig, false, bz,
                                                       (short)0, pc, false, false);
        // patch: rows of batch b0+1 need j=15,16 (rows s+32, s+33, outside K window)
        if (R0 + 15 >= 17 * (b0 + 1)) {    // statically elided for u = 0, 16
            const int b1 = b0 + 1;
            const float zl15 = (float)zsh[g * ZSTRIDE + 17 * b1 + 15];
            const float zl16 = (float)zsh[g * ZSTRIDE + 17 * b1 + 16];
#pragma unroll
            for (int r = 0; r < 8; ++r) {
                const int glo = R0 + r, ghi = glo + 8;
                const int blo = glo / 17, bhi = ghi / 17;        // constants
                const bool clo = (blo == b1), chi = (bhi == b1); // constants
                if (clo | chi) {
                    const int jlo = glo - 17 * blo, jhi = ghi - 17 * bhi;
                    int jj = hi ? jhi : jlo;
                    float add = aL[jj * 2 + 0] * zl15 + aL[jj * 2 + 1] * zl16;
                    bool cond = hi ? chi : clo;
                    d[r] += cond ? add : 0.0f;
                }
            }
        }
#pragma unroll
        for (int r = 0; r < 8; ++r) {
            int gr = R0 + r + (hi ? 8 : 0);
            outp[(size_t)gr * 128 + g] = d[r];
        }
    };

    // ---- main pipeline: stage1(t), then stage2(t-2) ----
    v8f pacc[3];
#pragma unroll
    for (int t = 0; t < 17; ++t) {
        // stage 1: GEMM for rows 16t..16t+15
        v8f acc0 = {}, acc1 = {};
        const int offbase = (16 * t + n) * XSTRIDE + (hi ? 8 : 0);
#pragma unroll
        for (int kt = 0; kt < 4; ++kt) {
            const h8* pa = (const h8*)&xsh[offbase + kt * 32];
            h8 alo = pa[0];
            h8 ahi = pa[2];   // +16 halves
            v16h a = __builtin_shufflevector(alo, ahi, 0, 1, 2, 3, 4, 5, 6, 7,
                                             8, 9, 10, 11, 12, 13, 14, 15);
            acc0 = __builtin_amdgcn_wmma_f32_16x16x32_f16(false, a, false, w0t[kt],
                                                          (short)0, acc0, false, false);
            acc1 = __builtin_amdgcn_wmma_f32_16x16x32_f16(false, a, false, w1t[kt],
                                                          (short)0, acc1, false, false);
        }
        // epilogue: z1 = M*h1 (f16 -> LDS), diag term + bias -> register ring.
        // j indices fold to constants selected by lane-half (no div-by-17 chains).
        h8 zst;
#pragma unroll
        for (int r = 0; r < 8; ++r) {
            const int jlo = (16 * t + r) % 17;        // constant
            const int jhi = (16 * t + r + 8) % 17;    // constant
            int jsel  = hi ? jhi : jlo;
            float mj  = Msh[jsel * 128 + g];
            float aj  = adiag[jsel];
            float mh0 = mj * acc0[r];
            float z1  = mj * acc1[r];
            zst[r] = (_Float16)z1;
            pacc[t % 3][r] = aj * (mh0 - z1) + biasv;
        }
        *(h8*)&zsh[g * ZSTRIDE + 16 * t + (hi ? 8 : 0)] = zst;

        if (t >= 2) stage2(t - 2, pacc[(t - 2) % 3]);
    }
    stage2(15, pacc[15 % 3]);
    stage2(16, pacc[16 % 3]);
}

extern "C" void kernel_launch(void* const* d_in, const int* in_sizes, int n_in,
                              void* d_out, int out_size, void* d_ws, size_t ws_size,
                              hipStream_t stream) {
    const float* x    = (const float*)d_in[0];
    const float* W    = (const float*)d_in[1];
    const float* Mmat = (const float*)d_in[2];
    const float* adj  = (const float*)d_in[3];
    const float* adj2 = (const float*)d_in[4];
    const float* bias = (const float*)d_in[5];
    float* out = (float*)d_out;

    int rows = in_sizes[0] / 128;   // B*J = 278528
    int nwg  = rows / ROWS;         // 1024 workgroups of 16 batches each

    mgcn_fused_wmma<<<dim3(nwg), dim3(256), 0, stream>>>(x, W, Mmat, adj, adj2, bias, out);
}
